// CostAwareHeteroMoE_77309411328331
// MI455X (gfx1250) — matrix operand order repaired and checked by reading, not compiled, not run on hardware
//
#include <hip/hip_runtime.h>
#include <hip/hip_bf16.h>

// ---------------------------------------------------------------------------
// CostAwareHeteroMoE forward for MI455X (gfx1250, wave32, WMMA).
// bf16 WMMA (16x16x32, f32 accum) tiled GEMMs; fp32->bf16 conversion during
// LDS staging.  Expert math factored as out = sum_e coef_e*(f_e - c_e) + sum c_e
// (gelu(0)=0 + gate normalization).  Staging is software-pipelined: next
// K-tile is held in registers while the current tile's 8 WMMAs issue.
// ---------------------------------------------------------------------------

typedef __attribute__((ext_vector_type(16))) __bf16 v16bf;
typedef __attribute__((ext_vector_type(8)))  __bf16 bf16x8;
typedef __attribute__((ext_vector_type(4)))  __bf16 bf16x4;
typedef __attribute__((ext_vector_type(8)))  float  v8f;

__device__ __forceinline__ float geluf(float v) {
    // exact erf GELU (matches jax.nn.gelu(approximate=False))
    return 0.5f * v * (1.0f + erff(v * 0.70710678118654752440f));
}

__device__ __forceinline__ float f4c(const float4& v, int i) {
    return i == 0 ? v.x : (i == 1 ? v.y : (i == 2 ? v.z : v.w));
}

enum EpiMode {
    EPI_GELU_STORE = 0,   // C = gelu(acc + bias)
    EPI_EXPERT_ACC = 1,   // C += coef[row*8] * (acc + bias - cvec)
    EPI_SHARED_FINAL = 2, // C = C + 0.1*(acc + bias) + cvec(=sum_e c_e)
    EPI_STORE_BIAS = 3,   // C = acc + bias
    EPI_ADD_BIAS = 4      // C += acc + bias
};

// ---------------------------------------------------------------------------
// Tiled bf16-WMMA GEMM: C[M,N] (+)= epi(A[M,K] @ B[K,N]).
// Block tile 128x128x32, 8 waves (4 row x 2 col), wave tile 32x64,
// 8 v_wmma_f32_16x16x32_bf16 per wave per K-step.  Register-level double
// buffering of the global tile (loads overlap WMMA).  All M,N,K multiples
// of 128 here -> no edge masking.
// ---------------------------------------------------------------------------
template <bool GELU_A, int EPI>
__global__ __launch_bounds__(256) void gemm_bf16_wmma(
    const float* __restrict__ A, const float* __restrict__ B,
    float* __restrict__ C, const float* __restrict__ bias,
    const float* __restrict__ coef,   // per-row gate coefficient base (stride 8) or null
    const float* __restrict__ cvec,   // per-col constant vector or null
    int M, int N, int K)
{
    constexpr int BM = 128, BN = 128, BK = 32, PAD = 8;
    __shared__ alignas(16) __bf16 As[BM][BK + PAD];   // [row][k]
    __shared__ alignas(16) __bf16 Bs[BN][BK + PAD];   // [col][k]

    const int t = threadIdx.x;
    const int rowBase = blockIdx.y * BM;
    const int colBase = blockIdx.x * BN;
    const int wave = t >> 5, lane = t & 31;
    const int wr = wave >> 1;          // 0..3  -> row offset wr*32
    const int wc = wave & 1;           // 0..1  -> col offset wc*64
    const int half = lane >> 4;        // lane half (ISA fragment layout)
    const int l16 = lane & 15;

    // staging geometry
    const int kbB = t >> 5;            // 0..7  : k-block (4 rows) for B transpose
    const int nbB = t & 31;            // 0..31 : n-block (4 cols) for B transpose

    v8f acc[2][4];
#pragma unroll
    for (int i = 0; i < 2; ++i)
#pragma unroll
        for (int j = 0; j < 4; ++j)
            acc[i][j] = (v8f){0.f, 0.f, 0.f, 0.f, 0.f, 0.f, 0.f, 0.f};

    float4 ra[4], rb[4];

    auto loadTile = [&](int k0) {
        // A: 4 float4 chunks, rows spread over the 128x32 tile
#pragma unroll
        for (int it = 0; it < 4; ++it) {
            int idx = it * 256 + t;            // 0..1023
            int r = idx >> 3;                  // 0..127
            int c4 = (idx & 7) << 2;           // 0..28
            ra[it] = *(const float4*)(A + (size_t)(rowBase + r) * K + k0 + c4);
        }
        // B: 4 consecutive K-rows, same 4-column group (for register transpose)
#pragma unroll
        for (int j = 0; j < 4; ++j)
            rb[j] = *(const float4*)(B + (size_t)(k0 + kbB * 4 + j) * N + colBase + nbB * 4);
    };

    auto storeTile = [&]() {
        // A -> LDS (optional prologue GELU), packed 4xbf16 = ds_store_b64
#pragma unroll
        for (int it = 0; it < 4; ++it) {
            int idx = it * 256 + t;
            int r = idx >> 3;
            int c4 = (idx & 7) << 2;
            float4 v = ra[it];
            if (GELU_A) { v.x = geluf(v.x); v.y = geluf(v.y); v.z = geluf(v.z); v.w = geluf(v.w); }
            bf16x4 p;
            p[0] = (__bf16)v.x; p[1] = (__bf16)v.y; p[2] = (__bf16)v.z; p[3] = (__bf16)v.w;
            *(bf16x4*)&As[r][c4] = p;
        }
        // B -> LDS transposed [n][k]; register transpose, ds_store_b64 per column
#pragma unroll
        for (int n = 0; n < 4; ++n) {
            bf16x4 q;
#pragma unroll
            for (int j = 0; j < 4; ++j) q[j] = (__bf16)f4c(rb[j], n);
            *(bf16x4*)&Bs[nbB * 4 + n][kbB * 4] = q;
        }
    };

    loadTile(0);

    for (int k0 = 0; k0 < K; k0 += BK) {
        storeTile();                       // waits on the in-flight global loads
        __syncthreads();

        const bool more = (k0 + BK) < K;
        if (more) loadTile(k0 + BK);       // next tile in flight during WMMA
        if (k0 + 2 * BK < K) {             // keep the HBM/L2 stream ahead
            __builtin_prefetch(A + (size_t)(rowBase + (t >> 3)) * K + (k0 + 2 * BK) + ((t & 7) << 2), 0, 1);
            __builtin_prefetch(B + (size_t)(k0 + 2 * BK + (t >> 5)) * N + colBase + ((t & 31) << 2), 0, 1);
        }

        // ---- fragments per CDNA5 16-bit layouts ----
        v16bf av[2], bv[4];
#pragma unroll
        for (int mi = 0; mi < 2; ++mi) {
            // A 16x32: lanes 0-15 hold K 0-7 / 16-23, lanes 16-31 hold K 8-15 / 24-31
            const __bf16* p = &As[wr * 32 + mi * 16 + l16][half * 8];
            bf16x8 lo = *(const bf16x8*)p;
            bf16x8 hi = *(const bf16x8*)(p + 16);
            av[mi] = __builtin_shufflevector(lo, hi, 0, 1, 2, 3, 4, 5, 6, 7,
                                             8, 9, 10, 11, 12, 13, 14, 15);
        }
#pragma unroll
        for (int ni = 0; ni < 4; ++ni) {
            // B 32x16: lanes 0-15 hold K 0-7 / 16-23, lanes 16-31 hold K 8-15 / 24-31
            const __bf16* p = &Bs[wc * 64 + ni * 16 + l16][half * 8];
            bf16x8 lo = *(const bf16x8*)p;
            bf16x8 hi = *(const bf16x8*)(p + 16);
            bv[ni] = __builtin_shufflevector(lo, hi, 0, 1, 2, 3, 4, 5, 6, 7,
                                             8, 9, 10, 11, 12, 13, 14, 15);
        }
#pragma unroll
        for (int mi = 0; mi < 2; ++mi)
#pragma unroll
            for (int ni = 0; ni < 4; ++ni)
                acc[mi][ni] = __builtin_amdgcn_wmma_f32_16x16x32_bf16(
                    false, av[mi], false, bv[ni], (short)0, acc[mi][ni], false, false);
        __syncthreads();
    }

    // ---- epilogue: C/D layout: VGPR r -> M = half*8 + r, l16 = column ----
#pragma unroll
    for (int mi = 0; mi < 2; ++mi) {
#pragma unroll
        for (int ni = 0; ni < 4; ++ni) {
            int col = colBase + wc * 64 + ni * 16 + l16;
#pragma unroll
            for (int r = 0; r < 8; ++r) {
                int row = rowBase + wr * 32 + mi * 16 + half * 8 + r;
                size_t idx = (size_t)row * N + col;
                float v = acc[mi][ni][r];
                if (EPI == EPI_GELU_STORE) {
                    C[idx] = geluf(v + bias[col]);
                } else if (EPI == EPI_EXPERT_ACC) {
                    C[idx] += coef[(size_t)row * 8] * (v + bias[col] - cvec[col]);
                } else if (EPI == EPI_SHARED_FINAL) {
                    C[idx] = C[idx] + 0.1f * (v + bias[col]) + cvec[col];
                } else if (EPI == EPI_STORE_BIAS) {
                    C[idx] = v + bias[col];
                } else { // EPI_ADD_BIAS
                    C[idx] += v + bias[col];
                }
            }
        }
    }
}

// ---------------------------------------------------------------------------
// Router: one wave32 per token. logits = x@rw + rb - 1e-7*cost; softmax;
// top-2 (stable, earliest index on ties); gate = softmax(top values);
// writes coef[token][8] = scatter of gate weights.
// ---------------------------------------------------------------------------
__global__ __launch_bounds__(256) void router_kernel(
    const float* __restrict__ x, const float* __restrict__ rw,
    const float* __restrict__ rb, float* __restrict__ coef)
{
    const int D = 2048;
    int token = blockIdx.x * 8 + (threadIdx.x >> 5);
    int lane = threadIdx.x & 31;
    const float* xr = x + (size_t)token * D;

    float acc[8] = {0.f, 0.f, 0.f, 0.f, 0.f, 0.f, 0.f, 0.f};
    for (int k = lane; k < D; k += 32) {
        float xv = xr[k];
        const float* rwk = rw + (size_t)k * 8;
#pragma unroll
        for (int e = 0; e < 8; ++e) acc[e] += xv * rwk[e];
    }
#pragma unroll
    for (int off = 16; off > 0; off >>= 1) {
#pragma unroll
        for (int e = 0; e < 8; ++e) acc[e] += __shfl_xor(acc[e], off, 32);
    }
    if (lane == 0) {
        const float hh[8] = {1024.f, 2048.f, 3072.f, 4096.f, 1024.f, 2048.f, 3072.f, 4096.f};
        float lg[8], mx = -3.4e38f;
#pragma unroll
        for (int e = 0; e < 8; ++e) {
            lg[e] = acc[e] + rb[e] - 1e-7f * (2.f * 1024.f * hh[e]);
            mx = fmaxf(mx, lg[e]);
        }
        float p[8], s = 0.f;
#pragma unroll
        for (int e = 0; e < 8; ++e) { p[e] = expf(lg[e] - mx); s += p[e]; }
        float inv = 1.f / s;
#pragma unroll
        for (int e = 0; e < 8; ++e) p[e] *= inv;
        int i0 = 0;
        for (int e = 1; e < 8; ++e) if (p[e] > p[i0]) i0 = e;
        int i1 = (i0 == 0) ? 1 : 0;
        for (int e = 0; e < 8; ++e) if (e != i0 && p[e] > p[i1]) i1 = e;
        float v0 = p[i0], v1 = p[i1];
        float m2 = fmaxf(v0, v1);
        float e0 = expf(v0 - m2), e1 = expf(v1 - m2);
        float g0 = e0 / (e0 + e1), g1 = e1 / (e0 + e1);
        float out[8] = {0.f, 0.f, 0.f, 0.f, 0.f, 0.f, 0.f, 0.f};
        out[i0] += g0; out[i1] += g1;
#pragma unroll
        for (int e = 0; e < 8; ++e) coef[(size_t)token * 8 + e] = out[e];
    }
}

// c_e[n] = gelu(b1_e) @ w2_e[:,n] + b2_e[n]   (per-expert "bias leak" constant)
__global__ __launch_bounds__(256) void cvec_kernel(
    const float* __restrict__ b1, const float* __restrict__ w2,
    const float* __restrict__ b2, float* __restrict__ cv, int h)
{
    __shared__ float gb[4096];
    for (int j = threadIdx.x; j < h; j += 256) gb[j] = geluf(b1[j]);
    __syncthreads();
    for (int c = threadIdx.x; c < 1024; c += 256) {
        float acc = b2[c];
        for (int j = 0; j < h; ++j) acc += gb[j] * w2[(size_t)j * 1024 + c];
        cv[c] = acc;
    }
}

__global__ __launch_bounds__(256) void csum_kernel(
    const float* __restrict__ cv, float* __restrict__ cs)
{
    int n = blockIdx.x * 256 + threadIdx.x;   // 1024 threads total
    float s = 0.f;
#pragma unroll
    for (int e = 0; e < 8; ++e) s += cv[(size_t)e * 1024 + n];
    cs[n] = s;
}

// ---------------------------------------------------------------------------
extern "C" void kernel_launch(void* const* d_in, const int* in_sizes, int n_in,
                              void* d_out, int out_size, void* d_ws, size_t ws_size,
                              hipStream_t stream)
{
    (void)in_sizes; (void)n_in; (void)out_size; (void)ws_size;

    const float* x        = (const float*)d_in[0];
    const float* down_w   = (const float*)d_in[1];
    const float* down_b   = (const float*)d_in[2];
    const float* up_w     = (const float*)d_in[3];
    const float* up_b     = (const float*)d_in[4];
    const float* router_w = (const float*)d_in[5];
    const float* router_b = (const float*)d_in[6];
    const float* sw1      = (const float*)d_in[7];
    const float* sb1      = (const float*)d_in[8];
    const float* sw2      = (const float*)d_in[9];
    const float* sb2      = (const float*)d_in[10];
    const float* core_w   = (const float*)d_in[11];
    const float* core_b   = (const float*)d_in[12];
    const float *ew1[8], *eb1[8], *ew2[8], *eb2[8];
    for (int e = 0; e < 8; ++e) {
        ew1[e] = (const float*)d_in[13 + 4 * e];
        eb1[e] = (const float*)d_in[14 + 4 * e];
        ew2[e] = (const float*)d_in[15 + 4 * e];
        eb2[e] = (const float*)d_in[16 + 4 * e];
    }

    constexpr int M = 8192, D = 2048, L = 1024;
    const int H[8] = {1024, 2048, 3072, 4096, 1024, 2048, 3072, 4096};

    // workspace layout (fp32): gh | s1 | moe | u(max h) | coef | cvec | csum
    float* ws   = (float*)d_ws;
    float* gh   = ws;
    float* s1   = gh + (size_t)M * L;
    float* moe  = s1 + (size_t)M * L;
    float* u    = moe + (size_t)M * L;
    float* coef = u + (size_t)M * 4096;
    float* cvec = coef + (size_t)M * 8;
    float* csum = cvec + (size_t)8 * L;

    dim3 blk(256);

    // 1) routing coefficients and per-expert constants
    router_kernel<<<M / 8, blk, 0, stream>>>(x, router_w, router_b, coef);
    for (int e = 0; e < 8; ++e)
        cvec_kernel<<<1, blk, 0, stream>>>(eb1[e], ew2[e], eb2[e], cvec + (size_t)e * L, H[e]);
    csum_kernel<<<4, blk, 0, stream>>>(cvec, csum);

    // 2) gh = gelu(x @ down_w + down_b)
    gemm_bf16_wmma<false, EPI_GELU_STORE><<<dim3(L / 128, M / 128), blk, 0, stream>>>(
        x, down_w, gh, down_b, nullptr, nullptr, M, L, D);

    // 3) experts: moe = sum_e coef_e * (f_e - c_e)
    hipMemsetAsync(moe, 0, (size_t)M * L * sizeof(float), stream);
    for (int e = 0; e < 8; ++e) {
        gemm_bf16_wmma<false, EPI_GELU_STORE><<<dim3(H[e] / 128, M / 128), blk, 0, stream>>>(
            gh, ew1[e], u, eb1[e], nullptr, nullptr, M, H[e], L);
        gemm_bf16_wmma<false, EPI_EXPERT_ACC><<<dim3(L / 128, M / 128), blk, 0, stream>>>(
            u, ew2[e], moe, eb2[e], coef + e, cvec + (size_t)e * L, M, L, H[e]);
    }

    // 4) shared path: moe += 0.1*(gelu(gh@sw1+sb1)@sw2 + sb2) + sum_e c_e
    gemm_bf16_wmma<false, EPI_GELU_STORE><<<dim3(L / 128, M / 128), blk, 0, stream>>>(
        gh, sw1, s1, sb1, nullptr, nullptr, M, L, L);
    gemm_bf16_wmma<false, EPI_SHARED_FINAL><<<dim3(L / 128, M / 128), blk, 0, stream>>>(
        s1, sw2, moe, sb2, nullptr, csum, M, L, L);

    // 5) out = moe @ up_w + up_b ; out += gelu(x) @ core_w + core_b
    gemm_bf16_wmma<false, EPI_STORE_BIAS><<<dim3(D / 128, M / 128), blk, 0, stream>>>(
        moe, up_w, (float*)d_out, up_b, nullptr, nullptr, M, D, L);
    gemm_bf16_wmma<true, EPI_ADD_BIAS><<<dim3(D / 128, M / 128), blk, 0, stream>>>(
        x, core_w, (float*)d_out, core_b, nullptr, nullptr, M, D, D);
}
